// EdgeGnn_13477607374967
// MI455X (gfx1250) — matrix operand order repaired and checked by reading, not compiled, not run on hardware
//
#include <hip/hip_runtime.h>
#include <hip/hip_bf16.h>
#include <stdint.h>
#include <stddef.h>

// ---------------------------------------------------------------------------
// EdgeGnn (depth=2) for MI455X / gfx1250.
// All heavy math -> v_wmma_f32_16x16x32_bf16 (bf16 inputs, f32 accumulate).
// v2: 64-row blocks (128KB LDS, legal on gfx1250's 320KB WGP LDS) and
// N-tile-major wave loop so each weight (B) fragment is loaded once per
// K-step and reused across 4 M-tiles in registers -> ~4x less L2 weight
// traffic vs v1, putting the kernel on the WMMA compute roofline.
// ---------------------------------------------------------------------------

#define N_NODES  16384
#define N_EDGES  131072
#define CCH      128
#define ROWS     64          // rows of the batch handled per block
#define MT       (ROWS / 16) // M-tiles per block (4)
#define THREADS  256         // 8 waves (wave32)
#define NWAVES   8
#define NEG_SLOPE 0.01f

typedef __bf16 bf16_t;
typedef __attribute__((ext_vector_type(16))) __bf16 v16bf;
typedef __attribute__((ext_vector_type(8)))  __bf16 v8bf;
typedef __attribute__((ext_vector_type(8)))  float  v8f;

// ---------------------------------------------------------------------------
// Utility kernels
// ---------------------------------------------------------------------------
__global__ void zero_f32_kernel(float* __restrict__ p, size_t n) {
  size_t i = (size_t)blockIdx.x * blockDim.x + threadIdx.x;
  if (i < n) p[i] = 0.f;
}

// W row-major [din][dout] fp32 -> Wt [doutP][dinP] bf16 (zero padded),
// bias fp32 [dout] -> padded fp32 [doutP].
__global__ void convert_linear_kernel(const float* __restrict__ W,
                                      const float* __restrict__ B,
                                      int din, int dout, int dinP, int doutP,
                                      bf16_t* __restrict__ Wt,
                                      float* __restrict__ bOut) {
  int i = blockIdx.x * blockDim.x + threadIdx.x;
  int n = i / dinP;
  int k = i - n * dinP;
  if (n < doutP) {
    float v = (k < din && n < dout) ? W[(size_t)k * dout + n] : 0.f;
    Wt[(size_t)n * dinP + k] = (bf16_t)v;
  }
  if (i < doutP) bOut[i] = (i < dout) ? B[i] : 0.f;
}

// Scatter-add of messages into per-node accumulator + degree count.
__global__ void scatter_add_kernel(const float* __restrict__ m,
                                   const int* __restrict__ dst,
                                   float* __restrict__ agg,
                                   float* __restrict__ cnt) {
  size_t i = (size_t)blockIdx.x * blockDim.x + threadIdx.x;
  if (i >= (size_t)N_EDGES * CCH) return;
  int e = (int)(i >> 7);       // /128
  int c = (int)(i & 127);
  int d = dst[e];
  atomicAdd(&agg[(size_t)d * CCH + c], m[i]);
  if (c == 0) atomicAdd(&cnt[d], 1.0f);
}

__global__ void mean_div_kernel(float* __restrict__ agg,
                                const float* __restrict__ cnt) {
  size_t i = (size_t)blockIdx.x * blockDim.x + threadIdx.x;
  if (i >= (size_t)N_NODES * CCH) return;
  agg[i] /= fmaxf(cnt[i >> 7], 1.0f);
}

// sym = 0.5*(fwd+bwd) written in-place into fwd; accumulate sum((fwd-bwd)^2).
__global__ void sym_loss_kernel(float* __restrict__ fwd,
                                const float* __restrict__ bwd,
                                float* __restrict__ sl) {
  __shared__ float red[THREADS];
  size_t i = (size_t)blockIdx.x * blockDim.x + threadIdx.x;
  float d2 = 0.f;
  if (i < (size_t)N_EDGES * CCH) {
    float f = fwd[i], b = bwd[i];
    fwd[i] = 0.5f * (f + b);
    float d = f - b;
    d2 = d * d;
  }
  red[threadIdx.x] = d2;
  __syncthreads();
  for (int s = THREADS / 2; s > 0; s >>= 1) {
    if ((int)threadIdx.x < s) red[threadIdx.x] += red[threadIdx.x + s];
    __syncthreads();
  }
  if (threadIdx.x == 0) atomicAdd(sl, red[0]);
}

__global__ void residual_add_kernel(const float* __restrict__ a,
                                    const float* __restrict__ b,
                                    float* __restrict__ o, size_t n) {
  size_t i = (size_t)blockIdx.x * blockDim.x + threadIdx.x;
  if (i < n) o[i] = a[i] + b[i];
}

__global__ void finalize_kernel(const float* __restrict__ sl,
                                float* __restrict__ out_scalar) {
  float denom = (float)N_EDGES * (float)CCH;
  *out_scalar = (sl[0] / denom + sl[1] / denom) * 0.5f;
}

// ---------------------------------------------------------------------------
// Fused 3-layer MLP with gather/concat staging.
//   in = concat(srcA[idxA[g]] (wA cols), srcB[idxB[g]] (wB cols), zeros pad)
//   out = L2( lrelu(L1( lrelu(L0(in)) )) )     (f32 to global)
// Weights: bf16 [dout][din] (pre-transposed, padded). All dims mult of 32.
// ---------------------------------------------------------------------------
__device__ __forceinline__ void stage_rows(bf16_t* __restrict__ lds, int d0,
                                           const float* __restrict__ srcA,
                                           const int* __restrict__ idxA, int wA,
                                           const float* __restrict__ srcB,
                                           const int* __restrict__ idxB, int wB,
                                           int row0, int R, int tid) {
  for (int i = tid; i < ROWS * d0; i += THREADS) {
    int r = i / d0;
    int c = i - r * d0;
    int g = row0 + r;
    float v = 0.f;
    if (g < R) {
      if (c < wA) {
        int s = idxA ? idxA[g] : g;
        v = srcA[(size_t)s * wA + c];
      } else if (c < wA + wB) {
        int s = idxB ? idxB[g] : g;
        v = srcB[(size_t)s * wB + (c - wA)];
      }
    }
    lds[i] = (bf16_t)v;
  }
}

// N-tile-major: each wave owns a set of output columns; per K-step the B
// (weight) fragment is loaded once and reused across all MT M-tiles.
__device__ __forceinline__ void mlp_layer(const bf16_t* __restrict__ ldsIn, int din,
                                          bf16_t* __restrict__ ldsOut, int dout,
                                          const bf16_t* __restrict__ Wt,
                                          const float* __restrict__ bias,
                                          bool lrelu,
                                          float* __restrict__ gOut,  // nullptr -> LDS out
                                          int row0, int R,
                                          int wave, int lane) {
  const int l15  = lane & 15;
  const int half = lane >> 4;
  const int aoff = half ? 16 : 0;
  const int ntiles = dout >> 4;
  for (int nt = wave; nt < ntiles; nt += NWAVES) {
    const int n = (nt << 4) + l15;
    const bf16_t* __restrict__ brow = Wt + (size_t)n * din + (half ? 16 : 0);
    v8f acc[MT];
#pragma unroll
    for (int mt = 0; mt < MT; ++mt)
      acc[mt] = (v8f){0.f, 0.f, 0.f, 0.f, 0.f, 0.f, 0.f, 0.f};
    for (int k0 = 0; k0 < din; k0 += 32) {
      // B fragment: 16 contiguous K for fixed N (weights pre-transposed)
      v16bf b = *(const v16bf*)(brow + k0);
#pragma unroll
      for (int mt = 0; mt < MT; ++mt) {
        // A fragment (contiguous K, matching B: lane<16 -> K 0..15; else 16..31)
        const bf16_t* arow = ldsIn + ((mt << 4) + l15) * din + aoff + k0;
        v16bf a = *(const v16bf*)(arow);
        acc[mt] = __builtin_amdgcn_wmma_f32_16x16x32_bf16(
            false, a, false, b, (short)0, acc[mt], false, false);
      }
    }
    const float bn = bias[n];
#pragma unroll
    for (int mt = 0; mt < MT; ++mt) {
#pragma unroll
      for (int r = 0; r < 8; ++r) {
        float v = acc[mt][r] + bn;
        if (lrelu) v = v > 0.f ? v : v * NEG_SLOPE;
        const int m = (mt << 4) + r + (half ? 8 : 0);
        if (gOut) {
          const int g = row0 + m;
          if (g < R) gOut[(size_t)g * dout + n] = v;
        } else {
          ldsOut[m * dout + n] = (bf16_t)v;
        }
      }
    }
  }
}

__global__ __launch_bounds__(THREADS)
void mlp3_kernel(const float* __restrict__ srcA, const int* __restrict__ idxA, int wA,
                 const float* __restrict__ srcB, const int* __restrict__ idxB, int wB,
                 int R,
                 const bf16_t* __restrict__ W0, const float* __restrict__ b0, int d0, int d1,
                 const bf16_t* __restrict__ W1, const float* __restrict__ b1, int d2,
                 const bf16_t* __restrict__ W2, const float* __restrict__ b2, int d3,
                 float* __restrict__ out) {
  __shared__ __align__(32) bf16_t lds0[ROWS * 512];   // 64KB
  __shared__ __align__(32) bf16_t lds1[ROWS * 512];   // 64KB (128KB total; <320KB WGP LDS)
  const int tid  = threadIdx.x;
  const int wave = tid >> 5;
  const int lane = tid & 31;
  const int row0 = blockIdx.x * ROWS;

  stage_rows(lds0, d0, srcA, idxA, wA, srcB, idxB, wB, row0, R, tid);
  __syncthreads();
  mlp_layer(lds0, d0, lds1, d1, W0, b0, true, nullptr, row0, R, wave, lane);
  __syncthreads();
  mlp_layer(lds1, d1, lds0, d2, W1, b1, true, nullptr, row0, R, wave, lane);
  __syncthreads();
  mlp_layer(lds0, d2, nullptr, d3, W2, b2, false, out, row0, R, wave, lane);
}

// ---------------------------------------------------------------------------
// Host side
// ---------------------------------------------------------------------------
static inline size_t align_up(size_t x, size_t a) { return (x + a - 1) & ~(a - 1); }
static inline int pad32(int x) { return (x + 31) & ~31; }

extern "C" void kernel_launch(void* const* d_in, const int* in_sizes, int n_in,
                              void* d_out, int out_size, void* d_ws, size_t ws_size,
                              hipStream_t stream) {
  (void)n_in; (void)out_size; (void)ws_size;
  const float* x0     = (const float*)d_in[0];            // [N, 128]
  const int*   eidx   = (const int*)d_in[1];              // [2, E] (int32 under default jax x64-off)
  const float* angles = (const float*)d_in[2];            // [E, 1]
  // d_in[3] = gt_edges (unused by reference forward)
  const int* src = eidx;
  const int* dst = eidx + N_EDGES;
  float* out = (float*)d_out;                             // E*128 edge feats + 1 scalar

  // Param leaves assumed in JAX pytree order (alphabetical dict keys):
  // ec1.out[0..2], ec1.pair[0..2], ec2.out, ec2.pair, nc1.msg, nc1.upd,
  // nc2.msg, nc2.upd; each linear contributes {b,w} (order resolved by size).
  enum { EC1_OUT = 0, EC1_PAIR = 3, EC2_OUT = 6, EC2_PAIR = 9,
         NC1_MSG = 12, NC1_UPD = 15, NC2_MSG = 18, NC2_UPD = 21 };
  static const int lin_din[24]  = {129,258,258, 256,512,512, 256,512,512, 256,512,512,
                                   256,512,512, 256,512,512, 256,512,512, 256,512,512};
  static const int lin_dout[24] = {258,258,128, 512,512,128, 512,512,128, 512,512,128,
                                   512,512,128, 512,512,128, 512,512,128, 512,512,128};

  const float* linW[24];
  const float* linB[24];
  int dinP[24], doutP[24];
  for (int i = 0; i < 24; ++i) {
    const int a = 4 + 2 * i, b = a + 1;
    if (in_sizes[a] >= in_sizes[b]) { linW[i] = (const float*)d_in[a]; linB[i] = (const float*)d_in[b]; }
    else                            { linW[i] = (const float*)d_in[b]; linB[i] = (const float*)d_in[a]; }
    dinP[i]  = pad32(lin_din[i]);
    doutP[i] = pad32(lin_dout[i]);
  }

  // ---- workspace carve-out (deterministic) ----
  char* ws = (char*)d_ws;
  size_t off = 0;
  bf16_t* wt[24]; float* bb[24];
  for (int i = 0; i < 24; ++i) {
    wt[i] = (bf16_t*)(ws + off); off = align_up(off + (size_t)doutP[i] * dinP[i] * 2, 256);
    bb[i] = (float*)(ws + off);  off = align_up(off + (size_t)doutP[i] * 4, 256);
  }
  const size_t NC = (size_t)N_NODES * CCH, EC = (size_t)N_EDGES * CCH;
  float* x1    = (float*)(ws + off); off = align_up(off + NC * 4, 256);
  float* x2    = (float*)(ws + off); off = align_up(off + NC * 4, 256);
  float* agg   = (float*)(ws + off); off = align_up(off + NC * 4, 256);
  float* ncout = (float*)(ws + off); off = align_up(off + NC * 4, 256);
  float* cnt   = (float*)(ws + off); off = align_up(off + (size_t)N_NODES * 4, 256);
  float* bufE1 = (float*)(ws + off); off = align_up(off + EC * 4, 256);
  float* bufE2 = (float*)(ws + off); off = align_up(off + EC * 4, 256);
  float* bufE3 = (float*)(ws + off); off = align_up(off + EC * 4, 256);
  float* sl    = (float*)(ws + off); off = align_up(off + 2 * 4, 256);

  // ---- weight conversion / transpose ----
  for (int i = 0; i < 24; ++i) {
    const int n = doutP[i] * dinP[i];
    convert_linear_kernel<<<(n + THREADS - 1) / THREADS, THREADS, 0, stream>>>(
        linW[i], linB[i], lin_din[i], lin_dout[i], dinP[i], doutP[i], wt[i], bb[i]);
  }
  zero_f32_kernel<<<1, 32, 0, stream>>>(sl, 2);

  auto launch_mlp3 = [&](const float* sa, const int* ia, int wa,
                         const float* sb, const int* ib, int wb,
                         int R, int L, float* o) {
    mlp3_kernel<<<R / ROWS, THREADS, 0, stream>>>(
        sa, ia, wa, sb, ib, wb, R,
        wt[L],     bb[L],     dinP[L],  doutP[L],
        wt[L + 1], bb[L + 1], doutP[L + 1],
        wt[L + 2], bb[L + 2], doutP[L + 2], o);
  };
  const int GE = (int)((EC + THREADS - 1) / THREADS);
  const int GN = (int)((NC + THREADS - 1) / THREADS);

  // ================= NC1 =================
  zero_f32_kernel<<<GN, THREADS, 0, stream>>>(agg, NC);
  zero_f32_kernel<<<(N_NODES + THREADS - 1) / THREADS, THREADS, 0, stream>>>(cnt, N_NODES);
  launch_mlp3(x0, dst, CCH, x0, src, CCH, N_EDGES, NC1_MSG, bufE1);          // m
  scatter_add_kernel<<<GE, THREADS, 0, stream>>>(bufE1, dst, agg, cnt);
  mean_div_kernel<<<GN, THREADS, 0, stream>>>(agg, cnt);
  launch_mlp3(x0, nullptr, CCH, agg, nullptr, CCH, N_NODES, NC1_UPD, x1);    // x1

  // ================= EC1 =================
  launch_mlp3(x1, src, CCH, x1, dst, CCH, N_EDGES, EC1_PAIR, bufE1);         // fwd
  launch_mlp3(x1, dst, CCH, x1, src, CCH, N_EDGES, EC1_PAIR, bufE2);         // bwd
  sym_loss_kernel<<<GE, THREADS, 0, stream>>>(bufE1, bufE2, &sl[0]);         // sym->bufE1
  launch_mlp3(bufE1, nullptr, CCH, angles, nullptr, 1, N_EDGES, EC1_OUT, bufE3); // e1

  // ================= NC2 =================
  zero_f32_kernel<<<GN, THREADS, 0, stream>>>(agg, NC);
  zero_f32_kernel<<<(N_NODES + THREADS - 1) / THREADS, THREADS, 0, stream>>>(cnt, N_NODES);
  launch_mlp3(x1, dst, CCH, x1, src, CCH, N_EDGES, NC2_MSG, bufE2);          // m
  scatter_add_kernel<<<GE, THREADS, 0, stream>>>(bufE2, dst, agg, cnt);
  mean_div_kernel<<<GN, THREADS, 0, stream>>>(agg, cnt);
  launch_mlp3(x1, nullptr, CCH, agg, nullptr, CCH, N_NODES, NC2_UPD, ncout);
  residual_add_kernel<<<GN, THREADS, 0, stream>>>(x1, ncout, x2, NC);        // x2

  // ================= EC2 =================
  launch_mlp3(x2, src, CCH, x2, dst, CCH, N_EDGES, EC2_PAIR, bufE1);         // fwd
  launch_mlp3(x2, dst, CCH, x2, src, CCH, N_EDGES, EC2_PAIR, bufE2);         // bwd
  sym_loss_kernel<<<GE, THREADS, 0, stream>>>(bufE1, bufE2, &sl[1]);         // sym->bufE1
  launch_mlp3(bufE1, nullptr, CCH, bufE3, nullptr, CCH, N_EDGES, EC2_OUT, out); // e2 -> d_out

  finalize_kernel<<<1, 1, 0, stream>>>(sl, out + EC);                        // side loss
}
